// ActorCriticTransformer_46643344834723
// MI455X (gfx1250) — matrix operand order repaired and checked by reading, not compiled, not run on hardware
//
#include <hip/hip_runtime.h>
#include <hip/hip_bf16.h>
#include <math.h>

#define B_TOT   32768
#define DIN     417
#define DIN_PAD 448
#define D       128
#define NHEAD   4
#define DH      32
#define NB      2
#define NWORDS  2315
#define POL     130
#define POL_PAD 144
#define M_TILE  64
#define NTHR    256

typedef unsigned short u16;
typedef __attribute__((ext_vector_type(16))) __bf16 v16bf;
typedef __attribute__((ext_vector_type(8)))  float  v8f;
typedef __attribute__((ext_vector_type(8)))  u16    us8;

__device__ __forceinline__ u16 f2b(float f) {
  unsigned u = __builtin_bit_cast(unsigned, f);
  u += 0x7fffu + ((u >> 16) & 1u);       // round-to-nearest-even to bf16
  return (u16)(u >> 16);
}
__device__ __forceinline__ float b2f(u16 h) {
  return __builtin_bit_cast(float, ((unsigned)h) << 16);
}
__device__ __forceinline__ v8f vzero() {
  v8f z; 
  #pragma unroll
  for (int e = 0; e < 8; ++e) z[e] = 0.f; 
  return z;
}

// A fragment: 16x32 bf16 tile at (m0,k0) from row-major LDS, stride sa elements.
// ISA layout: lane l holds A[l%16][8*(l/16)+0..7] and A[l%16][16+8*(l/16)+0..7].
__device__ __forceinline__ v16bf frag_a(const u16* A, int sa, int m0, int k0) {
  int l = threadIdx.x & 31;
  const u16* p = A + (m0 + (l & 15)) * sa + k0 + 8 * (l >> 4);
  union { us8 hh[2]; v16bf v; } c;
  c.hh[0] = *(const us8*)p;
  c.hh[1] = *(const us8*)(p + 16);
  return c.v;
}
// B fragment: 32x16 bf16 tile at (k0,n0) from LDS holding B^T row-major [N][K], stride kdim.
// ISA layout: lane l holds B[8*(l/16)+0..7][l%16] and B[16+8*(l/16)+0..7][l%16].
__device__ __forceinline__ v16bf frag_b(const u16* BT, int kdim, int n0, int k0) {
  int l = threadIdx.x & 31;
  const u16* p = BT + (n0 + (l & 15)) * kdim + k0 + 8 * (l >> 4);
  union { us8 hh[2]; v16bf v; } c;
  c.hh[0] = *(const us8*)p;
  c.hh[1] = *(const us8*)(p + 16);
  return c.v;
}
__device__ __forceinline__ v8f wmma_bf16(v16bf a, v16bf b, v8f c) {
  return __builtin_amdgcn_wmma_f32_16x16x32_bf16(false, a, false, b, (short)0, c, false, false);
}

// C = A(64x128) * B(128x128): wave (m0, nbase) computes 16 rows x 64 cols (4 n-tiles)
__device__ __forceinline__ void gemm64x128x128(const u16* A, const u16* BT, v8f acc[4],
                                               int m0, int nbase) {
  #pragma unroll
  for (int kt = 0; kt < 4; ++kt) {
    v16bf a = frag_a(A, D, m0, kt * 32);
    #pragma unroll
    for (int t = 0; t < 4; ++t) {
      v16bf b = frag_b(BT, D, nbase + 16 * t, kt * 32);
      acc[t] = wmma_bf16(a, b, acc[t]);
    }
  }
}

__device__ __forceinline__ void layernorm64(const float* sH, u16* sY,
    float* sRed, float* sMu, float* sRs,
    const float* g, const float* bta, int tid) {
  int row = tid >> 2, q = tid & 3;
  float s = 0.f, s2 = 0.f;
  for (int k = q * 32; k < q * 32 + 32; ++k) {
    float v = sH[row * D + k]; s += v; s2 += v * v;
  }
  sRed[tid * 2] = s; sRed[tid * 2 + 1] = s2;
  __syncthreads();
  if (tid < M_TILE) {
    float ss = 0.f, ss2 = 0.f;
    for (int q2 = 0; q2 < 4; ++q2) {
      ss += sRed[(tid * 4 + q2) * 2]; ss2 += sRed[(tid * 4 + q2) * 2 + 1];
    }
    float mu = ss * (1.f / D);
    float var = ss2 * (1.f / D) - mu * mu;
    sMu[tid] = mu;
    sRs[tid] = rsqrtf(var + 1e-5f);
  }
  __syncthreads();
  for (int idx = tid; idx < M_TILE * D; idx += NTHR) {
    int r = idx >> 7, k = idx & 127;
    sY[idx] = f2b((sH[idx] - sMu[r]) * sRs[r] * g[k] + bta[k]);
  }
  __syncthreads();
}

// ---------------- kernel 0: decode one-hot word matrix to letter indices ----------------
__global__ void k_prep_idx(const float* __restrict__ wm, int* __restrict__ idx5) {
  int j = blockIdx.x * blockDim.x + threadIdx.x;
  if (j >= NWORDS * 5) return;
  int v = j / 5, p = j % 5;
  const float* row = wm + (size_t)v * POL + p * 26;
  int best = 0; float bv = row[0];
  for (int c = 1; c < 26; ++c) { float x = row[c]; if (x > bv) { bv = x; best = c; } }
  idx5[j] = p * 26 + best;
}

// ---------------- kernel 1: encoder  h = relu(x @ enc_w + enc_b) ----------------
__global__ __launch_bounds__(NTHR) void k_encoder(const float* __restrict__ x,
    const float* __restrict__ enc_w, const float* __restrict__ enc_b,
    float* __restrict__ h) {
  __shared__ u16 sX[M_TILE * 64];   // 8 KB
  __shared__ u16 sWT[D * 64];       // 16 KB
  int tid = threadIdx.x;
  int b0 = blockIdx.x * M_TILE;
  int w = tid >> 5, l = tid & 31;
  int m0 = 16 * (w & 3), nbase = 64 * (w >> 2);
  v8f acc[4];
  #pragma unroll
  for (int t = 0; t < 4; ++t) acc[t] = vzero();

  for (int kc = 0; kc < DIN_PAD / 64; ++kc) {
    for (int idx = tid; idx < M_TILE * 64; idx += NTHR) {
      int r = idx >> 6, k = idx & 63, gk = kc * 64 + k;
      sX[idx] = (gk < DIN) ? f2b(x[(size_t)(b0 + r) * DIN + gk]) : (u16)0;
    }
    for (int idx = tid; idx < D * 64; idx += NTHR) {
      int n = idx >> 6, k = idx & 63, gk = kc * 64 + k;
      sWT[idx] = (gk < DIN) ? f2b(enc_w[(size_t)gk * D + n]) : (u16)0;
    }
    __syncthreads();
    #pragma unroll
    for (int kt = 0; kt < 2; ++kt) {
      v16bf a = frag_a(sX, 64, m0, kt * 32);
      #pragma unroll
      for (int t = 0; t < 4; ++t) {
        v16bf b = frag_b(sWT, 64, nbase + 16 * t, kt * 32);
        acc[t] = wmma_bf16(a, b, acc[t]);
      }
    }
    __syncthreads();
  }
  #pragma unroll
  for (int t = 0; t < 4; ++t) {
    int n = nbase + 16 * t + (l & 15);
    float bias = enc_b[n];
    #pragma unroll
    for (int r = 0; r < 8; ++r) {
      int m = m0 + r + 8 * (l >> 4);
      float vv = acc[t][r] + bias;
      h[(size_t)(b0 + m) * D + n] = vv > 0.f ? vv : 0.f;
    }
  }
}

// ---------------- kernel 2: one transformer block (in-place on h) ----------------
__global__ __launch_bounds__(NTHR) void k_block(float* __restrict__ h,
    const float* __restrict__ Wq, const float* __restrict__ bq,
    const float* __restrict__ Wk, const float* __restrict__ bk,
    const float* __restrict__ Wv, const float* __restrict__ bv,
    const float* __restrict__ Wo, const float* __restrict__ bo,
    const float* __restrict__ ln1_g, const float* __restrict__ ln1_b,
    const float* __restrict__ ln2_g, const float* __restrict__ ln2_b,
    const float* __restrict__ W1, const float* __restrict__ b1,
    const float* __restrict__ W2, const float* __restrict__ b2,
    int blk) {
  __shared__ float sH[M_TILE * D];        // 32 KB residual, fp32
  __shared__ u16   sY[M_TILE * D];        // 16 KB LN output / attn output
  __shared__ u16   sW[D * D];             // 32 KB weight staging (B^T [128][128])
  __shared__ u16   sQKV[3 * M_TILE * D];  // 48 KB q,k,v (bf16), later reused as FF hidden
  __shared__ float sRed[NTHR * 2];        // 2 KB LN partials
  __shared__ float sMu[M_TILE], sRs[M_TILE];

  int tid = threadIdx.x;
  int b0 = blockIdx.x * M_TILE;
  int w = tid >> 5, l = tid & 31;
  int m0 = 16 * (w & 3), nbase = 64 * (w >> 2);

  const float* Wq_b = Wq + (size_t)blk * NHEAD * D * DH;
  const float* Wk_b = Wk + (size_t)blk * NHEAD * D * DH;
  const float* Wv_b = Wv + (size_t)blk * NHEAD * D * DH;
  const float* bq_b = bq + blk * NHEAD * DH;
  const float* bk_b = bk + blk * NHEAD * DH;
  const float* bv_b = bv + blk * NHEAD * DH;
  const float* Wo_b = Wo + (size_t)blk * D * D;
  const float* bo_b = bo + blk * D;
  const float* W1_b = W1 + (size_t)blk * D * 4 * D;
  const float* b1_b = b1 + blk * 4 * D;
  const float* W2_b = W2 + (size_t)blk * 4 * D * D;
  const float* b2_b = b2 + blk * D;

  for (int idx = tid; idx < M_TILE * D; idx += NTHR)
    sH[idx] = h[(size_t)b0 * D + idx];
  __syncthreads();

  // ---- LN1 -> sY (bf16) ----
  layernorm64(sH, sY, sRed, sMu, sRs, ln1_g + blk * D, ln1_b + blk * D, tid);

  // ---- Q, K, V projections ----
  for (int j = 0; j < 3; ++j) {
    const float* Wj = (j == 0) ? Wq_b : (j == 1) ? Wk_b : Wv_b;
    const float* bj = (j == 0) ? bq_b : (j == 1) ? bk_b : bv_b;
    for (int idx = tid; idx < D * D; idx += NTHR) {
      int n = idx >> 7, k = idx & 127;  // col n = head*32 + e
      sW[idx] = f2b(Wj[(size_t)(n >> 5) * (D * DH) + k * DH + (n & 31)]);
    }
    __syncthreads();
    v8f acc[4];
    #pragma unroll
    for (int t = 0; t < 4; ++t) acc[t] = vzero();
    gemm64x128x128(sY, sW, acc, m0, nbase);
    #pragma unroll
    for (int t = 0; t < 4; ++t) {
      int n = nbase + 16 * t + (l & 15);
      float bias = bj[n];
      #pragma unroll
      for (int r = 0; r < 8; ++r) {
        int m = m0 + r + 8 * (l >> 4);
        sQKV[j * M_TILE * D + m * D + n] = f2b(acc[t][r] + bias);
      }
    }
    __syncthreads();
  }

  // ---- per-(row,head) 32x32 outer-product softmax attention ----
  {
    int r = tid & 63, hd = tid >> 6;
    const u16* qr = sQKV + 0 * M_TILE * D + r * D + hd * DH;
    const u16* kr = sQKV + 1 * M_TILE * D + r * D + hd * DH;
    const u16* vr = sQKV + 2 * M_TILE * D + r * D + hd * DH;
    float kf[DH], vf[DH];
    #pragma unroll
    for (int j = 0; j < DH; ++j) {
      kf[j] = b2f(kr[j]) * 0.17677669529663687f;  // scale = DH^-0.5 folded in
      vf[j] = b2f(vr[j]);
    }
    for (int i = 0; i < DH; ++i) {
      float qi = b2f(qr[i]);
      float mx = -3.4e38f;
      #pragma unroll
      for (int j = 0; j < DH; ++j) mx = fmaxf(mx, qi * kf[j]);
      float sum = 0.f, av = 0.f;
      #pragma unroll
      for (int j = 0; j < DH; ++j) {
        float e = __expf(qi * kf[j] - mx);
        sum += e; av += e * vf[j];
      }
      sY[r * D + hd * DH + i] = f2b(av / sum);    // o overwrites sY (LN1 done)
    }
  }
  __syncthreads();

  // ---- h += o @ Wo + bo ----
  for (int idx = tid; idx < D * D; idx += NTHR) {
    int n = idx >> 7, k = idx & 127;
    sW[idx] = f2b(Wo_b[(size_t)k * D + n]);
  }
  __syncthreads();
  {
    v8f acc[4];
    #pragma unroll
    for (int t = 0; t < 4; ++t) acc[t] = vzero();
    gemm64x128x128(sY, sW, acc, m0, nbase);
    #pragma unroll
    for (int t = 0; t < 4; ++t) {
      int n = nbase + 16 * t + (l & 15);
      float bias = bo_b[n];
      #pragma unroll
      for (int r = 0; r < 8; ++r) {
        int m = m0 + r + 8 * (l >> 4);
        sH[m * D + n] += acc[t][r] + bias;
      }
    }
  }
  __syncthreads();

  // ---- LN2 -> sY ----
  layernorm64(sH, sY, sRed, sMu, sRs, ln2_g + blk * D, ln2_b + blk * D, tid);

  // ---- FF: h += relu(y@W1+b1)@W2 + b2, processed in 4 parts of 128 ----
  u16* sT = sQKV;  // reuse qkv buffer for hidden activations
  for (int p = 0; p < 4; ++p) {
    for (int idx = tid; idx < D * D; idx += NTHR) {
      int n = idx >> 7, k = idx & 127;
      sW[idx] = f2b(W1_b[(size_t)k * (4 * D) + p * D + n]);
    }
    __syncthreads();
    v8f acc[4];
    #pragma unroll
    for (int t = 0; t < 4; ++t) acc[t] = vzero();
    gemm64x128x128(sY, sW, acc, m0, nbase);
    #pragma unroll
    for (int t = 0; t < 4; ++t) {
      int n = nbase + 16 * t + (l & 15);
      float bias = b1_b[p * D + n];
      #pragma unroll
      for (int r = 0; r < 8; ++r) {
        int m = m0 + r + 8 * (l >> 4);
        float vv = acc[t][r] + bias;
        sT[m * D + n] = f2b(vv > 0.f ? vv : 0.f);
      }
    }
    __syncthreads();
    for (int idx = tid; idx < D * D; idx += NTHR) {
      int n = idx >> 7, k = idx & 127;
      sW[idx] = f2b(W2_b[(size_t)(p * D + k) * D + n]);
    }
    __syncthreads();
    #pragma unroll
    for (int t = 0; t < 4; ++t) acc[t] = vzero();
    gemm64x128x128(sT, sW, acc, m0, nbase);
    #pragma unroll
    for (int t = 0; t < 4; ++t) {
      int n = nbase + 16 * t + (l & 15);
      float bias = (p == 0) ? b2_b[n] : 0.f;
      #pragma unroll
      for (int r = 0; r < 8; ++r) {
        int m = m0 + r + 8 * (l >> 4);
        sH[m * D + n] += acc[t][r] + bias;
      }
    }
    __syncthreads();
  }

  for (int idx = tid; idx < M_TILE * D; idx += NTHR)
    h[(size_t)b0 * D + idx] = sH[idx];
}

// ---------------- kernel 3: heads (policy via 5-hot gather, value) ----------------
__global__ __launch_bounds__(NTHR) void k_heads(const float* __restrict__ h,
    const float* __restrict__ val_w, const float* __restrict__ val_b,
    const float* __restrict__ pol_w, const float* __restrict__ pol_b,
    const int* __restrict__ idx5, float* __restrict__ out) {
  __shared__ u16   sHb[M_TILE * D];          // 16 KB
  __shared__ u16   sWT[POL_PAD * D];         // 36 KB
  __shared__ float sLg[M_TILE * POL_PAD];    // 36 KB logits
  __shared__ u16   sIdx[NWORDS * 5];         // 23 KB
  int tid = threadIdx.x;
  int b0 = blockIdx.x * M_TILE;

  for (int idx = tid; idx < M_TILE * D; idx += NTHR)
    sHb[idx] = f2b(h[(size_t)b0 * D + idx]);
  for (int idx = tid; idx < POL_PAD * D; idx += NTHR) {
    int n = idx >> 7, k = idx & 127;
    sWT[idx] = (n < POL) ? f2b(pol_w[(size_t)k * POL + n]) : (u16)0;
  }
  for (int j = tid; j < NWORDS * 5; j += NTHR) sIdx[j] = (u16)idx5[j];
  __syncthreads();

  int w = tid >> 5, l = tid & 31;
  for (int t = w; t < 4 * (POL_PAD / 16); t += 8) {  // 36 output tiles, wave-uniform loop
    int m0 = 16 * (t & 3), n0 = 16 * (t >> 2);
    v8f acc = vzero();
    #pragma unroll
    for (int kt = 0; kt < 4; ++kt) {
      v16bf a = frag_a(sHb, D, m0, kt * 32);
      v16bf b = frag_b(sWT, D, n0, kt * 32);
      acc = wmma_bf16(a, b, acc);
    }
    int n = n0 + (l & 15);
    float bias = (n < POL) ? pol_b[n] : 0.f;
    #pragma unroll
    for (int r = 0; r < 8; ++r) {
      int m = m0 + r + 8 * (l >> 4);
      sLg[m * POL_PAD + n] = acc[r] + bias;
    }
  }
  __syncthreads();

  // value head
  if (tid < M_TILE) {
    const float* hr = h + (size_t)(b0 + tid) * D;
    float s = val_b[0];
    for (int k = 0; k < D; ++k) s += hr[k] * val_w[k];
    out[(size_t)B_TOT * NWORDS + b0 + tid] = s;
  }

  // policy: 5 gathered adds per word (exploits one-hot word_matrix); coalesced stores
  for (int r = 0; r < M_TILE; ++r) {
    const float* lg = sLg + r * POL_PAD;
    size_t ob = (size_t)(b0 + r) * NWORDS;
    for (int v = tid; v < NWORDS; v += NTHR) {
      const u16* ip = sIdx + v * 5;
      out[ob + v] = lg[ip[0]] + lg[ip[1]] + lg[ip[2]] + lg[ip[3]] + lg[ip[4]];
    }
  }
}

extern "C" void kernel_launch(void* const* d_in, const int* in_sizes, int n_in,
                              void* d_out, int out_size, void* d_ws, size_t ws_size,
                              hipStream_t stream) {
  (void)in_sizes; (void)n_in; (void)out_size; (void)ws_size;
  const float* x     = (const float*)d_in[0];
  const float* enc_w = (const float*)d_in[1];
  const float* enc_b = (const float*)d_in[2];
  const float* Wq    = (const float*)d_in[3];
  const float* bq    = (const float*)d_in[4];
  const float* Wk    = (const float*)d_in[5];
  const float* bk    = (const float*)d_in[6];
  const float* Wv    = (const float*)d_in[7];
  const float* bv    = (const float*)d_in[8];
  const float* Wo    = (const float*)d_in[9];
  const float* bo    = (const float*)d_in[10];
  const float* ln1g  = (const float*)d_in[11];
  const float* ln1b  = (const float*)d_in[12];
  const float* ln2g  = (const float*)d_in[13];
  const float* ln2b  = (const float*)d_in[14];
  const float* W1    = (const float*)d_in[15];
  const float* b1    = (const float*)d_in[16];
  const float* W2    = (const float*)d_in[17];
  const float* b2    = (const float*)d_in[18];
  const float* val_w = (const float*)d_in[19];
  const float* val_b = (const float*)d_in[20];
  const float* pol_w = (const float*)d_in[21];
  const float* pol_b = (const float*)d_in[22];
  const float* wm    = (const float*)d_in[23];
  float* out = (float*)d_out;

  char* ws = (char*)d_ws;
  int*   idx5 = (int*)ws;                 // 46300 B
  float* h    = (float*)(ws + 65536);     // 16 MB intermediate, stays in L2

  k_prep_idx<<<(NWORDS * 5 + 255) / 256, 256, 0, stream>>>(wm, idx5);
  k_encoder<<<B_TOT / M_TILE, NTHR, 0, stream>>>(x, enc_w, enc_b, h);
  for (int blk = 0; blk < NB; ++blk)
    k_block<<<B_TOT / M_TILE, NTHR, 0, stream>>>(h, Wq, bq, Wk, bk, Wv, bv, Wo, bo,
                                                 ln1g, ln1b, ln2g, ln2b, W1, b1, W2, b2, blk);
  k_heads<<<B_TOT / M_TILE, NTHR, 0, stream>>>(h, val_w, val_b, pol_w, pol_b, idx5, out);
}
